// SparseTriangleSelfAttention_37134287242029
// MI455X (gfx1250) — matrix operand order repaired and checked by reading, not compiled, not run on hardware
//
#include <hip/hip_runtime.h>

#define N_CNT  1024
#define E_CNT  32768
#define EE_CNT 262144
#define C_S    384
#define C_Z    128
#define C_G    16
#define H_CNT  4
#define R_CNT  64

typedef __attribute__((ext_vector_type(16))) __bf16 v16bf;
typedef __attribute__((ext_vector_type(8)))  __bf16 v8bf;
typedef __attribute__((ext_vector_type(8)))  float  v8f;
typedef __attribute__((ext_vector_type(4)))  float  v4f;

// ---------------------------------------------------------------- helpers

__device__ __forceinline__ v8f zero8() {
  v8f z = {0.f, 0.f, 0.f, 0.f, 0.f, 0.f, 0.f, 0.f};
  return z;
}

__device__ __forceinline__ v16bf join16(v8bf lo, v8bf hi) {
  union {
    v16bf v;
    v8bf h[2];
  } u;
  u.h[0] = lo;
  u.h[1] = hi;
  return u.v;
}

// K index (within a 32-wide chunk) held by element e of the A fragment for
// this lane (CDNA5 ISA 7.12.2, 16-bit A-matrix 16x32 layout).
__device__ __forceinline__ int wmma_a_k(int lane, int e) {
  int base = (lane & 16) ? 8 : 0;
  return (e < 8) ? (base + e) : (base + 16 + (e - 8));
}

// B fragment from TRANSPOSED weights (layout [N, K] row-major, ld = K):
// lane holds column col0+(lane&15); its K-values are two runs of 8 (k = 8h+s, 16+8h+s).
// Two 128-bit loads per fragment.
__device__ __forceinline__ v16bf load_bT_bf16(const __bf16* __restrict__ p, int ldk,
                                              int k0, int col0, int lane) {
  int col = col0 + (lane & 15);
  int kb  = k0 + ((lane & 16) ? 8 : 0);
  const __bf16* pr = p + col * ldk + kb;
  v8bf lo = *(const v8bf*)pr;
  v8bf hi = *(const v8bf*)(pr + 16);
  return join16(lo, hi);
}

// A fragment from row-major bf16 [M, K]: element runs 0..7 and 8..15 are each
// contiguous in memory -> two 128-bit loads.
__device__ __forceinline__ v16bf load_a_bf16(const __bf16* __restrict__ p, int ld,
                                             int row0, int k0, int lane) {
  int row  = row0 + (lane & 15);
  int base = (lane & 16) ? 8 : 0;
  const __bf16* pr = p + row * ld + k0 + base;
  v8bf lo = *(const v8bf*)pr;
  v8bf hi = *(const v8bf*)(pr + 16);
  return join16(lo, hi);
}

// A fragment from row-major f32 [M, K]: 4x float4 loads + pack to bf16.
__device__ __forceinline__ v16bf load_a_f32(const float* __restrict__ p, int ld,
                                            int row0, int k0, int lane) {
  int row  = row0 + (lane & 15);
  int base = (lane & 16) ? 8 : 0;
  const float* pr = p + row * ld + k0 + base;
  v4f f0 = *(const v4f*)pr;
  v4f f1 = *(const v4f*)(pr + 4);
  v4f f2 = *(const v4f*)(pr + 16);
  v4f f3 = *(const v4f*)(pr + 20);
  v16bf a;
#pragma unroll
  for (int i = 0; i < 4; ++i) {
    a[i]      = (__bf16)f0[i];
    a[4 + i]  = (__bf16)f1[i];
    a[8 + i]  = (__bf16)f2[i];
    a[12 + i] = (__bf16)f3[i];
  }
  return a;
}

__device__ __forceinline__ v8f wmma_bf16(v16bf a, v16bf b, v8f c) {
  return __builtin_amdgcn_wmma_f32_16x16x32_bf16(false, a, false, b, (short)0, c,
                                                 false, false);
}

__device__ __forceinline__ float sigmoidf(float x) { return 1.f / (1.f + __expf(-x)); }

__device__ __forceinline__ void atomicMaxF(float* addr, float val) {
  unsigned int* ua = (unsigned int*)addr;
  unsigned int old = __float_as_uint(*addr);
  while (__uint_as_float(old) < val) {
    unsigned int assumed = old;
    old = atomicCAS(ua, assumed, __float_as_uint(val));
    if (old == assumed) break;
  }
}

// ---------------------------------------------------------------- kernels

// Convert f32 [K, N] -> bf16 transposed [N, K].
__global__ void transpose_to_bf16_kernel(const float* __restrict__ src,
                                         __bf16* __restrict__ dst, int K, int N) {
  int i = blockIdx.x * blockDim.x + threadIdx.x;
  if (i < K * N) {
    int k = i / N, n = i - k * N;
    dst[n * K + k] = (__bf16)src[i];
  }
}

// per-edge layernorm over C_Z=128 -> bf16.  One wave per row (4 rows/block).
__global__ __launch_bounds__(128) void ln_kernel(const float* __restrict__ ef,
                                                 const float* __restrict__ g,
                                                 const float* __restrict__ b,
                                                 __bf16* __restrict__ out) {
  int wid = threadIdx.x >> 5, lane = threadIdx.x & 31;
  int row = blockIdx.x * 4 + wid;
  float x[4];
  float s = 0.f;
#pragma unroll
  for (int i = 0; i < 4; ++i) {
    x[i] = ef[row * C_Z + lane + 32 * i];
    s += x[i];
  }
#pragma unroll
  for (int m = 16; m; m >>= 1) s += __shfl_xor(s, m, 32);
  float mean = s * (1.f / 128.f);
  float vs = 0.f;
#pragma unroll
  for (int i = 0; i < 4; ++i) {
    float t = x[i] - mean;
    vs += t * t;
  }
#pragma unroll
  for (int m = 16; m; m >>= 1) vs += __shfl_xor(vs, m, 32);
  float inv = rsqrtf(vs * (1.f / 128.f) + 1e-5f);
#pragma unroll
  for (int i = 0; i < 4; ++i) {
    int c = lane + 32 * i;
    out[row * C_Z + c] = (__bf16)((x[i] - mean) * inv * g[c] + b[c]);
  }
}

// nl/nr node projections: [1024,384] @ [384,16].  One wave per 16-row tile.
__global__ __launch_bounds__(128) void node_proj_kernel(
    const float* __restrict__ nf, const __bf16* __restrict__ WnlT,
    const __bf16* __restrict__ WnrT, const float* __restrict__ b_nl,
    const float* __restrict__ b_nr, float* __restrict__ nl_out,
    float* __restrict__ nr_out) {
  int wid = threadIdx.x >> 5, lane = threadIdx.x & 31;
  int row0 = (blockIdx.x * 4 + wid) * 16;
  v8f accl = zero8(), accr = zero8();
#pragma unroll
  for (int kc = 0; kc < 12; ++kc) {
    v16bf a = load_a_f32(nf, C_S, row0, kc * 32, lane);
    accl = wmma_bf16(a, load_bT_bf16(WnlT, C_S, kc * 32, 0, lane), accl);
    accr = wmma_bf16(a, load_bT_bf16(WnrT, C_S, kc * 32, 0, lane), accr);
  }
  int ccol = lane & 15, rofs = (lane & 16) ? 8 : 0;
#pragma unroll
  for (int r = 0; r < 8; ++r) {
    int row = row0 + r + rofs;
    nl_out[row * C_G + ccol] = accl[r] + b_nl[ccol];
    nr_out[row * C_G + ccol] = accr[r] + b_nr[ccol];
  }
}

// Fused q / k / v / sigmoid(og) projections from layernormed edge features.
// N-total = 512 (32 N-tiles); A fragments loaded once and reused.
__global__ __launch_bounds__(256) void edge_gemm_kernel(
    const __bf16* __restrict__ efb, const __bf16* __restrict__ WqT,
    const __bf16* __restrict__ WkvT, const __bf16* __restrict__ WogT,
    const float* __restrict__ b_q, const float* __restrict__ b_kv,
    const float* __restrict__ b_og, float* __restrict__ q, float* __restrict__ k,
    float* __restrict__ v, float* __restrict__ ogs) {
  int wid = threadIdx.x >> 5, lane = threadIdx.x & 31;
  int row0 = (blockIdx.x * 8 + wid) * 16;
  v16bf afrag[4];
#pragma unroll
  for (int kc = 0; kc < 4; ++kc) afrag[kc] = load_a_bf16(efb, C_Z, row0, kc * 32, lane);
  int ccol = lane & 15, rofs = (lane & 16) ? 8 : 0;
  for (int nt = 0; nt < 32; ++nt) {
    const __bf16* W;
    const float* bp;
    float* dst;
    int cn, dcol;
    bool sig = false;
    if (nt < 8) {
      W = WqT; cn = nt * 16; dst = q; dcol = cn; bp = b_q + cn;
    } else if (nt < 16) {
      W = WkvT; cn = (nt - 8) * 16; dst = k; dcol = cn; bp = b_kv + cn;
    } else if (nt < 24) {
      W = WkvT; cn = C_Z + (nt - 16) * 16; dst = v; dcol = cn - C_Z; bp = b_kv + cn;
    } else {
      W = WogT; cn = (nt - 24) * 16; dst = ogs; dcol = cn; bp = b_og + cn;
      sig = true;
    }
    v8f acc = zero8();
#pragma unroll
    for (int kc = 0; kc < 4; ++kc)
      acc = wmma_bf16(afrag[kc], load_bT_bf16(W, C_Z, kc * 32, cn, lane), acc);
    float bb = bp[ccol];
#pragma unroll
    for (int r = 0; r < 8; ++r) {
      float x = acc[r] + bb;
      if (sig) x = sigmoidf(x);
      dst[(row0 + r + rofs) * C_Z + dcol + ccol] = x;
    }
  }
}

__global__ void init_kernel(float* __restrict__ upd, float* __restrict__ segmax,
                            float* __restrict__ segsum) {
  int i = blockIdx.x * blockDim.x + threadIdx.x;
  if (i < E_CNT * C_Z) upd[i] = 0.f;
  if (i < E_CNT * H_CNT) {
    segmax[i] = -3.0e38f;
    segsum[i] = 0.f;
  }
}

// Heavy kernel: per 16 edge-edge pairs build A = outer(nl[n1], nr[n2]) (K=256)
// and A = rbf(dist) (K=64) in registers, WMMA against W_bg / W_db, fuse
// sigmoid(gate)*dist_bias, reduce @ W_tb via LDS, add q.k logits, atomicMax.
__global__ __launch_bounds__(128) void pair_kernel(
    const int* __restrict__ eei, const int* __restrict__ edge_index,
    const float* __restrict__ node_trans, const float* __restrict__ nl,
    const float* __restrict__ nr, const __bf16* __restrict__ WbgT,
    const float* __restrict__ b_bg, const __bf16* __restrict__ WdbT,
    const float* __restrict__ b_db, const float* __restrict__ Wtb,
    const float* __restrict__ qbuf, const float* __restrict__ kbuf,
    float* __restrict__ attn, float* __restrict__ segmax) {
  __shared__ __align__(16) float s_nl[4][16][16];
  __shared__ __align__(16) float s_nr[4][16][16];
  __shared__ __align__(16) float s_sb[4][16][16];
  __shared__ float s_d[4][16];
  __shared__ int s_e0[4][16];
  __shared__ int s_e1[4][16];

  int wid = threadIdx.x >> 5, lane = threadIdx.x & 31;
  int tile = blockIdx.x * 4 + wid;
  int row0 = tile * 16;

  // Phase 1: gather per-row operands into LDS (split across half-waves).
  if (lane < 16) {
    int rr = lane;
    int ee0 = eei[row0 + rr];
    int ee1 = eei[EE_CNT + row0 + rr];
    s_e0[wid][rr] = ee0;
    s_e1[wid][rr] = ee1;
    int n1 = edge_index[ee1];
    int n2 = edge_index[ee0];
    const v4f* src = (const v4f*)(nl + n1 * C_G);
    v4f* dst = (v4f*)&s_nl[wid][rr][0];
#pragma unroll
    for (int i = 0; i < 4; ++i) dst[i] = src[i];
    float dx = node_trans[n1 * 3 + 0] - node_trans[n2 * 3 + 0] + 1e-8f;
    float dy = node_trans[n1 * 3 + 1] - node_trans[n2 * 3 + 1] + 1e-8f;
    float dz = node_trans[n1 * 3 + 2] - node_trans[n2 * 3 + 2] + 1e-8f;
    s_d[wid][rr] = sqrtf(dx * dx + dy * dy + dz * dz);
  } else {
    int rr = lane - 16;
    int ee0 = eei[row0 + rr];
    int n2 = edge_index[ee0];
    const v4f* src = (const v4f*)(nr + n2 * C_G);
    v4f* dst = (v4f*)&s_nr[wid][rr][0];
#pragma unroll
    for (int i = 0; i < 4; ++i) dst[i] = src[i];
  }
  __syncthreads();

  int arow = lane & 15;
  int ccol = lane & 15;
  int rofs = (lane & 16) ? 8 : 0;

  // Build outer-product A fragments (K = 256, 8 chunks) once.
  v16bf agate[8];
#pragma unroll
  for (int kc = 0; kc < 8; ++kc) {
#pragma unroll
    for (int e = 0; e < 16; ++e) {
      int kk = kc * 32 + wmma_a_k(lane, e);
      agate[kc][e] = (__bf16)(s_nl[wid][arow][kk >> 4] * s_nr[wid][arow][kk & 15]);
    }
  }
  // RBF A fragments (K = 64, 2 chunks): exp(-((d-mu_k)/sigma)^2).
  float d = s_d[wid][arow];
  const float mu_step = 20.0f / 63.0f;   // linspace(0,20,64) spacing
  const float inv_sigma = 64.0f / 20.0f; // sigma = 20/64
  v16bf arbf[2];
#pragma unroll
  for (int kc = 0; kc < 2; ++kc) {
#pragma unroll
    for (int e = 0; e < 16; ++e) {
      int kk = kc * 32 + wmma_a_k(lane, e);
      float t = (d - mu_step * (float)kk) * inv_sigma;
      arbf[kc][e] = (__bf16)__expf(-t * t);
    }
  }

  // Each lane owns two (row, head) bias accumulators: combos lane and lane+32.
  int row_a = lane >> 2;
  int row_b = 8 + (lane >> 2);
  int h_a = lane & 3;
  float bias0 = 0.f, bias1 = 0.f;

  for (int nt = 0; nt < 8; ++nt) {
    int cn = nt * 16;
    v8f gacc = zero8();
#pragma unroll
    for (int kc = 0; kc < 8; ++kc)
      gacc = wmma_bf16(agate[kc], load_bT_bf16(WbgT, 2 * C_Z, kc * 32, cn, lane), gacc);
    v8f dacc = zero8();
#pragma unroll
    for (int kc = 0; kc < 2; ++kc)
      dacc = wmma_bf16(arbf[kc], load_bT_bf16(WdbT, R_CNT, kc * 32, cn, lane), dacc);
    float bgc = b_bg[cn + ccol];
    float bdc = b_db[cn + ccol];
#pragma unroll
    for (int r = 0; r < 8; ++r)
      s_sb[wid][r + rofs][ccol] = sigmoidf(gacc[r] + bgc) * (dacc[r] + bdc);
    __syncthreads();
    float p0 = 0.f, p1 = 0.f;
#pragma unroll
    for (int c = 0; c < 16; ++c) {
      float wtb = Wtb[(cn + c) * H_CNT + h_a];
      p0 += s_sb[wid][row_a][c] * wtb;
      p1 += s_sb[wid][row_b][c] * wtb;
    }
    bias0 += p0;
    bias1 += p1;
    __syncthreads();
  }

  // attention logits + segment max (vectorized q.k gathers).
  const float scale = 0.08838834764831845f; // 1/sqrt(128)
  {
    int e0 = s_e0[wid][row_a], e1 = s_e1[wid][row_a];
    const v4f* qp = (const v4f*)(qbuf + e1 * C_Z + h_a * 32);
    const v4f* kp = (const v4f*)(kbuf + e0 * C_Z + h_a * 32);
    float acc = 0.f;
#pragma unroll
    for (int t = 0; t < 8; ++t) {
      v4f qa = qp[t], ka = kp[t];
#pragma unroll
      for (int j = 0; j < 4; ++j) acc += qa[j] * ka[j];
    }
    float a = acc * scale + bias0;
    attn[(row0 + row_a) * H_CNT + h_a] = a;
    atomicMaxF(&segmax[e1 * H_CNT + h_a], a);
  }
  {
    int e0 = s_e0[wid][row_b], e1 = s_e1[wid][row_b];
    const v4f* qp = (const v4f*)(qbuf + e1 * C_Z + h_a * 32);
    const v4f* kp = (const v4f*)(kbuf + e0 * C_Z + h_a * 32);
    float acc = 0.f;
#pragma unroll
    for (int t = 0; t < 8; ++t) {
      v4f qa = qp[t], ka = kp[t];
#pragma unroll
      for (int j = 0; j < 4; ++j) acc += qa[j] * ka[j];
    }
    float a = acc * scale + bias1;
    attn[(row0 + row_b) * H_CNT + h_a] = a;
    atomicMaxF(&segmax[e1 * H_CNT + h_a], a);
  }
}

__global__ void softmax_p_kernel(const int* __restrict__ eei,
                                 float* __restrict__ attn,
                                 const float* __restrict__ segmax,
                                 float* __restrict__ segsum) {
  int i = blockIdx.x * blockDim.x + threadIdx.x;
  if (i >= EE_CNT * H_CNT) return;
  int pair = i >> 2, h = i & 3;
  int e1 = eei[EE_CNT + pair];
  float p = __expf(attn[i] - segmax[e1 * H_CNT + h]);
  attn[i] = p;
  atomicAdd(&segsum[e1 * H_CNT + h], p);
}

__global__ void scatter_upd_kernel(const int* __restrict__ eei,
                                   const float* __restrict__ attn,
                                   const float* __restrict__ segsum,
                                   const float* __restrict__ v,
                                   float* __restrict__ upd) {
  int i = blockIdx.x * blockDim.x + threadIdx.x;
  if (i >= EE_CNT * C_Z) return;
  int pair = i >> 7, c = i & 127, h = c >> 5;
  int e0 = eei[pair];
  int e1 = eei[EE_CNT + pair];
  float w = attn[pair * H_CNT + h] / (segsum[e1 * H_CNT + h] + 1e-16f);
  atomicAdd(&upd[e1 * C_Z + c], v[e0 * C_Z + c] * w);
}

// out = (upd * sigmoid(og)) @ W_out + b_out
__global__ __launch_bounds__(256) void out_kernel(
    const float* __restrict__ upd, const float* __restrict__ ogs,
    const __bf16* __restrict__ WoutT, const float* __restrict__ b_out,
    float* __restrict__ out) {
  int wid = threadIdx.x >> 5, lane = threadIdx.x & 31;
  int row0 = (blockIdx.x * 8 + wid) * 16;
  int arow = row0 + (lane & 15);
  int base = (lane & 16) ? 8 : 0;
  v16bf afrag[4];
#pragma unroll
  for (int kc = 0; kc < 4; ++kc) {
    const float* pu = upd + arow * C_Z + kc * 32 + base;
    const float* po = ogs + arow * C_Z + kc * 32 + base;
    v4f u0 = *(const v4f*)pu, u1 = *(const v4f*)(pu + 4);
    v4f u2 = *(const v4f*)(pu + 16), u3 = *(const v4f*)(pu + 20);
    v4f o0 = *(const v4f*)po, o1 = *(const v4f*)(po + 4);
    v4f o2 = *(const v4f*)(po + 16), o3 = *(const v4f*)(po + 20);
#pragma unroll
    for (int i = 0; i < 4; ++i) {
      afrag[kc][i]      = (__bf16)(u0[i] * o0[i]);
      afrag[kc][4 + i]  = (__bf16)(u1[i] * o1[i]);
      afrag[kc][8 + i]  = (__bf16)(u2[i] * o2[i]);
      afrag[kc][12 + i] = (__bf16)(u3[i] * o3[i]);
    }
  }
  int ccol = lane & 15, rofs = (lane & 16) ? 8 : 0;
  for (int nt = 0; nt < 8; ++nt) {
    int cn = nt * 16;
    v8f acc = zero8();
#pragma unroll
    for (int kc = 0; kc < 4; ++kc)
      acc = wmma_bf16(afrag[kc], load_bT_bf16(WoutT, C_Z, kc * 32, cn, lane), acc);
    float bb = b_out[cn + ccol];
#pragma unroll
    for (int r = 0; r < 8; ++r)
      out[(row0 + r + rofs) * C_Z + cn + ccol] = acc[r] + bb;
  }
}

// ---------------------------------------------------------------- launcher

extern "C" void kernel_launch(void* const* d_in, const int* in_sizes, int n_in,
                              void* d_out, int out_size, void* d_ws, size_t ws_size,
                              hipStream_t stream) {
  const float* node_features = (const float*)d_in[0];
  const float* node_trans    = (const float*)d_in[1];
  const float* edge_features = (const float*)d_in[2];
  const int*   edge_index    = (const int*)d_in[3];
  const int*   eei           = (const int*)d_in[4];
  const float* ln_g  = (const float*)d_in[5];
  const float* ln_b  = (const float*)d_in[6];
  const float* W_nl  = (const float*)d_in[7];
  const float* b_nl  = (const float*)d_in[8];
  const float* W_nr  = (const float*)d_in[9];
  const float* b_nr  = (const float*)d_in[10];
  const float* W_bg  = (const float*)d_in[11];
  const float* b_bg  = (const float*)d_in[12];
  const float* W_db  = (const float*)d_in[13];
  const float* b_db  = (const float*)d_in[14];
  const float* W_tb  = (const float*)d_in[15];
  const float* W_q   = (const float*)d_in[16];
  const float* b_q   = (const float*)d_in[17];
  const float* W_kv  = (const float*)d_in[18];
  const float* b_kv  = (const float*)d_in[19];
  const float* W_og  = (const float*)d_in[20];
  const float* b_og  = (const float*)d_in[21];
  const float* W_out = (const float*)d_in[22];
  const float* b_out = (const float*)d_in[23];
  float* out = (float*)d_out;

  char* wsb = (char*)d_ws;
  size_t off = 0;
  auto take = [&](size_t bytes) -> char* {
    char* p = wsb + off;
    off += (bytes + 255) & ~(size_t)255;
    return p;
  };

  __bf16* ef_bf   = (__bf16*)take((size_t)E_CNT * C_Z * 2);
  float*  qb      = (float*)take((size_t)E_CNT * C_Z * 4);
  float*  kb      = (float*)take((size_t)E_CNT * C_Z * 4);
  float*  vb      = (float*)take((size_t)E_CNT * C_Z * 4);
  float*  ogs     = (float*)take((size_t)E_CNT * C_Z * 4);
  float*  updb    = (float*)take((size_t)E_CNT * C_Z * 4);
  float*  attnb   = (float*)take((size_t)EE_CNT * H_CNT * 4);
  float*  segmaxb = (float*)take((size_t)E_CNT * H_CNT * 4);
  float*  segsumb = (float*)take((size_t)E_CNT * H_CNT * 4);
  float*  nlb     = (float*)take((size_t)N_CNT * C_G * 4);
  float*  nrb     = (float*)take((size_t)N_CNT * C_G * 4);
  // transposed bf16 weights: [N, K] layout
  __bf16* WnlT  = (__bf16*)take((size_t)C_S * C_G * 2);
  __bf16* WnrT  = (__bf16*)take((size_t)C_S * C_G * 2);
  __bf16* WbgT  = (__bf16*)take((size_t)C_G * C_G * C_Z * 2);
  __bf16* WdbT  = (__bf16*)take((size_t)R_CNT * C_Z * 2);
  __bf16* WqT   = (__bf16*)take((size_t)C_Z * C_Z * 2);
  __bf16* WkvT  = (__bf16*)take((size_t)C_Z * 2 * C_Z * 2);
  __bf16* WogT  = (__bf16*)take((size_t)C_Z * C_Z * 2);
  __bf16* WoutT = (__bf16*)take((size_t)C_Z * C_Z * 2);

  auto cvtT = [&](const float* s, __bf16* d, int K, int N) {
    transpose_to_bf16_kernel<<<(K * N + 255) / 256, 256, 0, stream>>>(s, d, K, N);
  };
  cvtT(W_nl, WnlT, C_S, C_G);
  cvtT(W_nr, WnrT, C_S, C_G);
  cvtT(W_bg, WbgT, C_G * C_G, C_Z);
  cvtT(W_db, WdbT, R_CNT, C_Z);
  cvtT(W_q, WqT, C_Z, C_Z);
  cvtT(W_kv, WkvT, C_Z, 2 * C_Z);
  cvtT(W_og, WogT, C_Z, C_Z);
  cvtT(W_out, WoutT, C_Z, C_Z);

  ln_kernel<<<E_CNT / 4, 128, 0, stream>>>(edge_features, ln_g, ln_b, ef_bf);

  node_proj_kernel<<<N_CNT / 64, 128, 0, stream>>>(node_features, WnlT, WnrT, b_nl,
                                                   b_nr, nlb, nrb);

  edge_gemm_kernel<<<E_CNT / 128, 256, 0, stream>>>(ef_bf, WqT, WkvT, WogT, b_q,
                                                    b_kv, b_og, qb, kb, vb, ogs);

  init_kernel<<<(E_CNT * C_Z + 255) / 256, 256, 0, stream>>>(updb, segmaxb, segsumb);

  pair_kernel<<<EE_CNT / 64, 128, 0, stream>>>(eei, edge_index, node_trans, nlb, nrb,
                                               WbgT, b_bg, WdbT, b_db, W_tb, qb, kb,
                                               attnb, segmaxb);

  softmax_p_kernel<<<(EE_CNT * H_CNT + 255) / 256, 256, 0, stream>>>(eei, attnb,
                                                                     segmaxb, segsumb);

  scatter_upd_kernel<<<(EE_CNT * C_Z + 255) / 256, 256, 0, stream>>>(eei, attnb,
                                                                     segsumb, vb, updb);

  out_kernel<<<E_CNT / 128, 256, 0, stream>>>(updb, ogs, WoutT, b_out, out);
}
